// Net_23192823399034
// MI455X (gfx1250) — matrix operand order repaired and checked by reading, not compiled, not run on hardware
//
#include <hip/hip_runtime.h>

// ---------------------------------------------------------------------------
// MI455X (gfx1250) implementation.
//  - Heavy 160-ch 3x3x3 convs + 512->160 pointwise run as implicit-GEMM on
//    v_wmma_f32_16x16x32_f16 (f16 in, f32 acc), channels-last f16 activations
//    with zero halos so every A/B fragment is a contiguous 32B v16h load.
//  - Register blocking: each wave owns 16 out-ch x 64 pixels (4 N-tiles),
//    reusing the A (weight) fragment 4x -> 1.25 vmem instr per WMMA instead
//    of 2, so the XDL pipe (not vmem issue) is the limiter.
//  - Small per-view convs, warp mask, cost-volume gather (K=81) stay on f32
//    VALU (<10% of FLOPs, shaped badly for 16x16 tiles).
//  - All tensors fit in the 192MB L2 -> negligible HBM traffic at 23.3TB/s.
// ---------------------------------------------------------------------------

typedef __attribute__((ext_vector_type(16))) _Float16 v16h;
typedef __attribute__((ext_vector_type(8)))  float    v8f;

#define ANG   9
#define HH    64
#define WW    64
#define NPIX  4096           // 64*64
#define NA    81             // 9*9
#define NDISP 9
#define CH    160            // trunk channels
#define MT    10             // 160/16 M-tiles
#define NROWS 576            // 9 * 64 rows; each wave does one row (4 N-tiles)
#define EPS   1e-5f

// padded activation buffer: 11 x 66 x 66 x 160 f16
#define PD 11
#define PH 66
#define PW 66
#define PADN ((size_t)PD * PH * PW * CH)

__device__ __forceinline__ float lrelu(float v) { return v >= 0.f ? v : 0.1f * v; }
__device__ __forceinline__ float bnap(float v, const float* p, int C, int c) {
    // p = [gamma(C), beta(C), mean(C), var(C)]
    return (v - p[2 * C + c]) * p[c] * rsqrtf(p[3 * C + c] + EPS) + p[C + c];
}

// ---------------------------------------------------------------------------
// Stage A helpers (f32 scalar)
// ---------------------------------------------------------------------------

__global__ __launch_bounds__(256) void k_rearrange(const float* __restrict__ x,
                                                   float* __restrict__ x5) {
    int idx = blockIdx.x * blockDim.x + threadIdx.x;
    if (idx >= NA * NPIX) return;
    int wq = idx & 63, hq = (idx >> 6) & 63, a = idx >> 12;
    int u = a / 9, v = a % 9;
    x5[idx] = x[(size_t)(u * 64 + hq) * 576 + v * 64 + wq];
}

// generic per-view 3x3 conv, [Cin][81][64][64] f32 -> [Cout][81][64][64]
__global__ __launch_bounds__(256) void k_convA(const float* __restrict__ in, int Cin,
                                               const float* __restrict__ w,
                                               const float* __restrict__ bias,
                                               const float* __restrict__ bnp,
                                               int Cout, int act,
                                               const float* __restrict__ resid,
                                               float* __restrict__ out) {
    int idx = blockIdx.x * blockDim.x + threadIdx.x;
    if (idx >= Cout * NA * NPIX) return;
    int wq = idx & 63, hq = (idx >> 6) & 63;
    int a  = (idx >> 12) % NA;
    int co = idx / (NA * NPIX);
    float acc = 0.f;
    for (int ci = 0; ci < Cin; ++ci) {
        const float* ip  = in + ((size_t)ci * NA + a) * NPIX;
        const float* wp9 = w + ((size_t)co * Cin + ci) * 9;
        #pragma unroll
        for (int kh = 0; kh < 3; ++kh) {
            int hh = hq + kh - 1;
            if (hh < 0 || hh >= HH) continue;
            #pragma unroll
            for (int kw = 0; kw < 3; ++kw) {
                int ww2 = wq + kw - 1;
                if (ww2 < 0 || ww2 >= WW) continue;
                acc += wp9[kh * 3 + kw] * ip[hh * 64 + ww2];
            }
        }
    }
    if (bias) acc += bias[co];
    if (bnp)  acc = bnap(acc, bnp, Cout, co);
    if (act)  acc = lrelu(acc);
    if (resid) acc += resid[idx];
    out[idx] = acc;
}

// warp mask: one thread per (view, pixel)
__global__ __launch_bounds__(256) void k_mask(const float* __restrict__ x,
                                              const float* __restrict__ disp,
                                              float* __restrict__ mask) {
    int idx = blockIdx.x * blockDim.x + threadIdx.x;
    if (idx >= NA * NPIX) return;
    int wq = idx & 63, hq = (idx >> 6) & 63, a = idx >> 12;
    int u = a / 9, v = a % 9;
    float d  = -disp[hq * 64 + wq];
    float px = ((float)wq / 63.0f + (float)(v - 4) * d / 64.0f) * 64.0f - 0.5f;
    float py = ((float)hq / 63.0f + (float)(u - 4) * d / 64.0f) * 64.0f - 0.5f;
    float x0 = floorf(px), y0 = floorf(py);
    float dx = px - x0, dy = py - y0;
    int x0i = (int)x0, y0i = (int)y0;
    auto tap = [&](int yi, int xi) -> float {
        if (xi < 0 || xi >= WW || yi < 0 || yi >= HH) return 0.f;
        return x[(size_t)(u * 64 + yi) * 576 + v * 64 + xi];
    };
    float wa = tap(y0i, x0i)     * (1.f - dx) * (1.f - dy)
             + tap(y0i, x0i + 1) * dx         * (1.f - dy)
             + tap(y0i + 1, x0i) * (1.f - dx) * dy
             + tap(y0i + 1, x0i + 1) * dx     * dy;
    float ref  = x[(size_t)(4 * 64 + hq) * 576 + 4 * 64 + wq];
    float diff = fabsf(wa - ref);
    if (u == 4 && v == 4) diff = 0.f;
    float m = 1.f - diff;
    mask[idx] = m * m;
}

__global__ __launch_bounds__(256) void k_maskmean(const float* __restrict__ mask,
                                                  float* __restrict__ mm) {
    int p = blockIdx.x * blockDim.x + threadIdx.x;
    if (p >= NPIX) return;
    float s = 0.f;
    for (int a = 0; a < NA; ++a) s += mask[(size_t)a * NPIX + p];
    mm[p] = s / (float)NA;
}

// cost volume: out channels-last f16 [(d*64+h)*64+w][512]
__global__ __launch_bounds__(256) void k_cost(const float* __restrict__ feat,
                                              const float* __restrict__ mask,
                                              const float* __restrict__ mm,
                                              const float* __restrict__ fw,
                                              _Float16* __restrict__ cost16) {
    int idx = blockIdx.x * blockDim.x + threadIdx.x;
    if (idx >= 512 * NDISP * NPIX) return;
    int wq = idx & 63, hq = (idx >> 6) & 63;
    int dd = (idx >> 12) % NDISP;
    int co = idx / (NDISP * NPIX);
    int g  = co >> 6;
    int ds = dd - 4;
    float acc = 0.f;
    for (int u = 0; u < 9; ++u) {
        int rr = hq + (4 - u) * ds;
        if (rr < 0 || rr >= HH) continue;
        for (int v = 0; v < 9; ++v) {
            int cc = wq + (4 - v) * ds;
            if (cc < 0 || cc >= WW) continue;
            int a = u * 9 + v;
            acc += fw[(size_t)co * NA + a]
                 * feat[((size_t)g * NA + a) * NPIX + rr * 64 + cc]
                 * mask[(size_t)a * NPIX + hq * 64 + wq];
        }
    }
    acc /= mm[hq * 64 + wq];
    cost16[((size_t)(dd * 64 + hq) * 64 + wq) * 512 + co] = (_Float16)acc;
}

__global__ __launch_bounds__(256) void k_zero16(_Float16* __restrict__ p, size_t n) {
    size_t i = (size_t)blockIdx.x * blockDim.x + threadIdx.x;
    if (i < n) p[i] = (_Float16)0.f;
}

// ---------------------------------------------------------------------------
// WMMA implicit GEMM path
// ---------------------------------------------------------------------------

// pack OIDHW f32 weights into per-lane A fragments (A 16x32, contiguous K order: K = hi*16 + element)
__global__ __launch_bounds__(256) void k_prepack(const float* __restrict__ w,
                                                 int I, int ntaps,
                                                 _Float16* __restrict__ wpk) {
    int idx = blockIdx.x * blockDim.x + threadIdx.x;
    int nkc = I >> 5;
    if (idx >= MT * ntaps * nkc * 32) return;
    int lane = idx & 31;
    int t    = idx >> 5;
    int kc   = t % nkc;  t /= nkc;
    int tap  = t % ntaps;
    int mt   = t / ntaps;
    int m  = mt * 16 + (lane & 15);
    int hi = lane >> 4;
    _Float16* dst = wpk + (size_t)idx * 16;
    #pragma unroll
    for (int j = 0; j < 8; ++j)       // halves 0..7  -> K = hi*16 + j
        dst[j] = (_Float16)w[((size_t)m * I + kc * 32 + hi * 16 + j) * ntaps + tap];
    #pragma unroll
    for (int j = 0; j < 8; ++j)       // halves 8..15 -> K = hi*16 + 8 + j
        dst[8 + j] = (_Float16)w[((size_t)m * I + kc * 32 + hi * 16 + 8 + j) * ntaps + tap];
}

// One wave computes a 16(out-ch) x 64(full w row, 4 N-tiles) strip.
// A fragment loaded once per (tap,kchunk) and reused by 4 WMMAs.
__global__ __launch_bounds__(256) void k_wmma_conv(
        const _Float16* __restrict__ apad, int Hp, int Wp, int Cin,
        int KD, int KH, int KW,
        const _Float16* __restrict__ wpk,
        const float* __restrict__ bnp, int act,
        float* __restrict__ out32,
        _Float16* __restrict__ opad, int OPHp, int OPWp, int opoff) {
    int gid  = blockIdx.x * blockDim.x + threadIdx.x;
    int wave = gid >> 5, lane = gid & 31;
    int mt = wave / NROWS;
    int nt = wave % NROWS;
    int d  = nt >> 6;                       // disparity slice
    int h  = nt & 63;                       // row
    int ln    = lane & 15;                  // N column within a 16-tile
    int khalf = (lane >> 4) << 4;           // B fragment: K = khalf + j
    int nkc   = Cin >> 5;
    int ntaps = KD * KH * KW;
    const size_t tstep = (size_t)16 * Cin;  // halves between adjacent N-tiles

    v8f acc0 = {0.f,0.f,0.f,0.f,0.f,0.f,0.f,0.f};
    v8f acc1 = acc0, acc2 = acc0, acc3 = acc0;

    int tap = 0;
    for (int kd = 0; kd < KD; ++kd)
        for (int kh = 0; kh < KH; ++kh)
            for (int kw = 0; kw < KW; ++kw, ++tap) {
                const _Float16* bb = apad
                    + ((size_t)((d + kd) * Hp + (h + kh)) * Wp + (ln + kw)) * (size_t)Cin
                    + khalf;
                const _Float16* ab = wpk
                    + (((size_t)(mt * ntaps + tap) * nkc) * 32 + lane) * 16;
                for (int kc = 0; kc < nkc; ++kc) {
                    v16h a  = *(const v16h*)(ab + (size_t)kc * 512);
                    const _Float16* bk = bb + kc * 32;
                    v16h b0 = *(const v16h*)(bk);
                    v16h b1 = *(const v16h*)(bk + tstep);
                    v16h b2 = *(const v16h*)(bk + 2 * tstep);
                    v16h b3 = *(const v16h*)(bk + 3 * tstep);
                    acc0 = __builtin_amdgcn_wmma_f32_16x16x32_f16(false, a, false, b0, (short)0, acc0, false, false);
                    acc1 = __builtin_amdgcn_wmma_f32_16x16x32_f16(false, a, false, b1, (short)0, acc1, false, false);
                    acc2 = __builtin_amdgcn_wmma_f32_16x16x32_f16(false, a, false, b2, (short)0, acc2, false, false);
                    acc3 = __builtin_amdgcn_wmma_f32_16x16x32_f16(false, a, false, b3, (short)0, acc3, false, false);
                }
            }

    // epilogue: lane holds 8 consecutive channels for 4 pixels (one per tile)
    int c0 = mt * 16 + ((lane >> 4) << 3);
    v8f accs[4] = {acc0, acc1, acc2, acc3};
    #pragma unroll
    for (int t2 = 0; t2 < 4; ++t2) {
        int col = t2 * 16 + ln;
        size_t n  = (size_t)(d * 64 + h) * 64 + col;
        size_t op = ((size_t)(d + opoff) * OPHp + (h + opoff)) * OPWp + (col + opoff);
        #pragma unroll
        for (int r = 0; r < 8; ++r) {
            int c = c0 + r;
            float v = accs[t2][r];
            if (bnp) v = bnap(v, bnp, CH, c);
            if (act) v = lrelu(v);
            if (out32) out32[n * CH + c] = v;
            if (opad)  opad[op * CH + c] = (_Float16)v;
        }
    }
}

// ---------------------------------------------------------------------------
// Channel attention + head
// ---------------------------------------------------------------------------

__global__ __launch_bounds__(256) void k_careduce(const float* __restrict__ t,
                                                  float* __restrict__ mean) {
    int c = blockIdx.x % CH;
    int d = blockIdx.x / CH;
    __shared__ float sm[256];
    float s = 0.f;
    for (int p = threadIdx.x; p < NPIX; p += 256)
        s += t[((size_t)d * NPIX + p) * CH + c];
    sm[threadIdx.x] = s;
    __syncthreads();
    for (int st = 128; st > 0; st >>= 1) {
        if (threadIdx.x < st) sm[threadIdx.x] += sm[threadIdx.x + st];
        __syncthreads();
    }
    if (threadIdx.x == 0) mean[d * CH + c] = sm[0] / (float)NPIX;
}

__global__ void k_camlp(const float* __restrict__ mean,
                        const float* __restrict__ w1, const float* __restrict__ b1,
                        const float* __restrict__ bn1,
                        const float* __restrict__ w2, const float* __restrict__ b2,
                        const float* __restrict__ bn2,
                        float* __restrict__ cay) {
    int d = blockIdx.x;
    int c = threadIdx.x;           // 160 threads
    __shared__ float y1[10];
    if (c < 10) {
        float s = b1[c];
        for (int k = 0; k < CH; ++k) s += w1[c * CH + k] * mean[d * CH + k];
        s = bnap(s, bn1, 10, c);
        y1[c] = lrelu(s);
    }
    __syncthreads();
    float s = b2[c];
    for (int k = 0; k < 10; ++k) s += w2[c * 10 + k] * y1[k];
    s = bnap(s, bn2, CH, c);
    cay[d * CH + c] = 1.f / (1.f + expf(-s));
}

__global__ __launch_bounds__(256) void k_cacombine(const float* __restrict__ t,
                                                   const float* __restrict__ cay,
                                                   float* __restrict__ bufA,
                                                   _Float16* __restrict__ opad) {
    int idx = blockIdx.x * blockDim.x + threadIdx.x;
    if (idx >= NDISP * NPIX * CH) return;
    int c = idx % CH;
    int n = idx / CH;
    int d = n >> 12, p = n & 4095, h = p >> 6, w = p & 63;
    float v = t[idx] * cay[d * CH + c] + bufA[idx];
    bufA[idx] = v;
    opad[(((size_t)(d + 1) * PH + (h + 1)) * PW + (w + 1)) * CH + c] = (_Float16)v;
}

__global__ __launch_bounds__(256) void k_c4(const _Float16* __restrict__ apad,
                                            const float* __restrict__ w,
                                            float* __restrict__ score) {
    int idx = blockIdx.x * blockDim.x + threadIdx.x;
    if (idx >= NDISP * NPIX) return;
    int p = idx & 4095, d = idx >> 12, h = p >> 6, wq = p & 63;
    float acc = 0.f;
    for (int kd = 0; kd < 3; ++kd)
        for (int kh = 0; kh < 3; ++kh)
            for (int kw = 0; kw < 3; ++kw) {
                int tap = (kd * 3 + kh) * 3 + kw;
                const _Float16* ip = apad
                    + (((size_t)(d + kd) * PH + (h + kh)) * PW + (wq + kw)) * CH;
                for (int ci = 0; ci < CH; ++ci)
                    acc += (float)ip[ci] * w[ci * 27 + tap];
            }
    score[idx] = acc;
}

__global__ __launch_bounds__(256) void k_softmax(const float* __restrict__ score,
                                                 float* __restrict__ out) {
    int p = blockIdx.x * blockDim.x + threadIdx.x;
    if (p >= NPIX) return;
    float m = -1e30f;
    for (int d = 0; d < NDISP; ++d) m = fmaxf(m, score[d * NPIX + p]);
    float s = 0.f, acc = 0.f;
    for (int d = 0; d < NDISP; ++d) {
        float e = expf(score[d * NPIX + p] - m);
        s += e;
        acc += e * (float)(d - 4);
    }
    out[p] = acc / s;
}

// ---------------------------------------------------------------------------
// Host orchestration
// ---------------------------------------------------------------------------

static inline int g1(long long n) { return (int)((n + 255) / 256); }

extern "C" void kernel_launch(void* const* d_in, const int* in_sizes, int n_in,
                              void* d_out, int out_size, void* d_ws, size_t ws_size,
                              hipStream_t stream) {
    (void)in_sizes; (void)n_in; (void)out_size; (void)ws_size;
    const float* x      = (const float*)d_in[0];
    const float* dispGT = (const float*)d_in[1];
    const float* if_w0  = (const float*)d_in[2];
    const float* if_bn0 = (const float*)d_in[3];
    const float* resb_w = (const float*)d_in[4];
    const float* resb_b = (const float*)d_in[5];
    const float* resb_bn= (const float*)d_in[6];
    const float* if_w1  = (const float*)d_in[7];
    const float* if_bn1 = (const float*)d_in[8];
    const float* if_w2  = (const float*)d_in[9];
    const float* if_bn2 = (const float*)d_in[10];
    const float* last_w = (const float*)d_in[11];
    const float* fuse_w = (const float*)d_in[12];
    const float* sq_w   = (const float*)d_in[13];
    const float* sq_bn  = (const float*)d_in[14];
    const float* c1_w   = (const float*)d_in[15];
    const float* c1_bn  = (const float*)d_in[16];
    const float* c2_w   = (const float*)d_in[17];
    const float* c2_bn  = (const float*)d_in[18];
    const float* c3_w   = (const float*)d_in[19];
    const float* c3_bn  = (const float*)d_in[20];
    const float* c4_w   = (const float*)d_in[21];
    const float* r3_w   = (const float*)d_in[22];
    const float* r3_bn  = (const float*)d_in[23];
    const float* ca_w1  = (const float*)d_in[24];
    const float* ca_b1  = (const float*)d_in[25];
    const float* ca_bn1 = (const float*)d_in[26];
    const float* ca_w2  = (const float*)d_in[27];
    const float* ca_b2  = (const float*)d_in[28];
    const float* ca_bn2 = (const float*)d_in[29];

    char* ws = (char*)d_ws;
    size_t off = 0;
    auto take = [&](size_t bytes) { char* p = ws + off; off += (bytes + 255) & ~(size_t)255; return p; };
    float*     X5     = (float*)take((size_t)NA * NPIX * 4);
    float*     HB     = (float*)take((size_t)16 * NA * NPIX * 4);
    float*     TB     = (float*)take((size_t)16 * NA * NPIX * 4);
    float*     FEAT   = (float*)take((size_t)8 * NA * NPIX * 4);
    float*     MASK   = (float*)take((size_t)NA * NPIX * 4);
    float*     MMEAN  = (float*)take((size_t)NPIX * 4);
    _Float16*  COST16 = (_Float16*)take((size_t)NDISP * NPIX * 512 * 2);
    _Float16*  APAD0  = (_Float16*)take(PADN * 2);
    _Float16*  APAD1  = (_Float16*)take(PADN * 2);
    float*     BUFA   = (float*)take((size_t)NDISP * NPIX * CH * 4);
    float*     BUFT   = (float*)take((size_t)NDISP * NPIX * CH * 4);
    _Float16*  WPK    = (_Float16*)take((size_t)CH * 4320 * 2);
    float*     SCORE  = (float*)take((size_t)NDISP * NPIX * 4);
    float*     CAMEAN = (float*)take((size_t)NDISP * CH * 4);
    float*     CAY    = (float*)take((size_t)NDISP * CH * 4);

    // -------- Stage A: per-view feature extraction (f32 VALU) --------
    k_rearrange<<<g1(NA * NPIX), 256, 0, stream>>>(x, X5);
    k_convA<<<g1(16LL * NA * NPIX), 256, 0, stream>>>(X5, 1, if_w0, nullptr, if_bn0, 16, 0, nullptr, HB);
    for (int i = 0; i < 8; ++i) {
        const float* w0 = resb_w + (size_t)(i * 2 + 0) * 16 * 16 * 9;
        const float* w1 = resb_w + (size_t)(i * 2 + 1) * 16 * 16 * 9;
        k_convA<<<g1(16LL * NA * NPIX), 256, 0, stream>>>(HB, 16, w0, resb_b + (i * 2) * 16,
                                                          resb_bn + (i * 2) * 64, 16, 1, nullptr, TB);
        k_convA<<<g1(16LL * NA * NPIX), 256, 0, stream>>>(TB, 16, w1, resb_b + (i * 2 + 1) * 16,
                                                          resb_bn + (i * 2 + 1) * 64, 16, 0, HB, HB);
    }
    k_convA<<<g1(16LL * NA * NPIX), 256, 0, stream>>>(HB, 16, if_w1, nullptr, if_bn1, 16, 1, nullptr, TB);
    k_convA<<<g1(8LL  * NA * NPIX), 256, 0, stream>>>(TB, 16, if_w2, nullptr, if_bn2, 8, 1, nullptr, HB);
    k_convA<<<g1(8LL  * NA * NPIX), 256, 0, stream>>>(HB, 8, last_w, nullptr, nullptr, 8, 0, nullptr, FEAT);

    // -------- Mask + cost volume --------
    k_mask<<<g1(NA * NPIX), 256, 0, stream>>>(x, dispGT, MASK);
    k_maskmean<<<g1(NPIX), 256, 0, stream>>>(MASK, MMEAN);
    k_cost<<<g1(512LL * NDISP * NPIX), 256, 0, stream>>>(FEAT, MASK, MMEAN, fuse_w, COST16);

    // -------- WMMA trunk --------
    k_zero16<<<g1((long long)PADN), 256, 0, stream>>>(APAD0, PADN);
    k_zero16<<<g1((long long)PADN), 256, 0, stream>>>(APAD1, PADN);
    const int WBLK = (MT * NROWS * 32) / 256;   // 720 blocks, exact (EXEC all ones)

    // sq: 512 -> 160 pointwise, BN + lrelu -> APAD0
    k_prepack<<<g1(MT * 1 * 16 * 32), 256, 0, stream>>>(sq_w, 512, 1, WPK);
    k_wmma_conv<<<WBLK, 256, 0, stream>>>(COST16, 64, 64, 512, 1, 1, 1, WPK, sq_bn, 1,
                                          nullptr, APAD0, PH, PW, 1);
    // c1
    k_prepack<<<g1(MT * 27 * 5 * 32), 256, 0, stream>>>(c1_w, CH, 27, WPK);
    k_wmma_conv<<<WBLK, 256, 0, stream>>>(APAD0, PH, PW, CH, 3, 3, 3, WPK, c1_bn, 1,
                                          nullptr, APAD1, PH, PW, 1);
    // c2 (also keep f32 residual base in BUFA)
    k_prepack<<<g1(MT * 27 * 5 * 32), 256, 0, stream>>>(c2_w, CH, 27, WPK);
    k_wmma_conv<<<WBLK, 256, 0, stream>>>(APAD1, PH, PW, CH, 3, 3, 3, WPK, c2_bn, 1,
                                          BUFA, APAD0, PH, PW, 1);
    // r3 blocks
    for (int i = 0; i < 2; ++i) {
        const float* wA  = r3_w  + (size_t)(i * 2 + 0) * CH * CH * 27;
        const float* wB  = r3_w  + (size_t)(i * 2 + 1) * CH * CH * 27;
        const float* bnA = r3_bn + (size_t)(i * 2 + 0) * 4 * CH;
        const float* bnB = r3_bn + (size_t)(i * 2 + 1) * 4 * CH;
        k_prepack<<<g1(MT * 27 * 5 * 32), 256, 0, stream>>>(wA, CH, 27, WPK);
        k_wmma_conv<<<WBLK, 256, 0, stream>>>(APAD0, PH, PW, CH, 3, 3, 3, WPK, bnA, 1,
                                              nullptr, APAD1, PH, PW, 1);
        k_prepack<<<g1(MT * 27 * 5 * 32), 256, 0, stream>>>(wB, CH, 27, WPK);
        k_wmma_conv<<<WBLK, 256, 0, stream>>>(APAD1, PH, PW, CH, 3, 3, 3, WPK, bnB, 0,
                                              BUFT, nullptr, PH, PW, 1);
        k_careduce<<<CH * NDISP, 256, 0, stream>>>(BUFT, CAMEAN);
        k_camlp<<<NDISP, CH, 0, stream>>>(CAMEAN,
                                          ca_w1 + (size_t)i * 10 * CH, ca_b1 + i * 10,
                                          ca_bn1 + i * 40,
                                          ca_w2 + (size_t)i * CH * 10, ca_b2 + i * CH,
                                          ca_bn2 + (size_t)i * 4 * CH, CAY);
        k_cacombine<<<g1((long long)NDISP * NPIX * CH), 256, 0, stream>>>(BUFT, CAY, BUFA, APAD0);
    }
    // c3
    k_prepack<<<g1(MT * 27 * 5 * 32), 256, 0, stream>>>(c3_w, CH, 27, WPK);
    k_wmma_conv<<<WBLK, 256, 0, stream>>>(APAD0, PH, PW, CH, 3, 3, 3, WPK, c3_bn, 1,
                                          nullptr, APAD1, PH, PW, 1);
    // head
    k_c4<<<g1(NDISP * NPIX), 256, 0, stream>>>(APAD1, c4_w, SCORE);
    k_softmax<<<g1(NPIX), 256, 0, stream>>>(SCORE, (float*)d_out);
}
